// global_graph_creator_5574867550489
// MI455X (gfx1250) — matrix operand order repaired and checked by reading, not compiled, not run on hardware
//
#include <hip/hip_runtime.h>
#include <hip/hip_bf16.h>

typedef __attribute__((ext_vector_type(16))) _Float16 v16h;
typedef __attribute__((ext_vector_type(8)))  _Float16 v8h;
typedef __attribute__((ext_vector_type(8)))  float    v8f;

#define N_ROWS 8192
#define DIM    128
#define KDIM   256     // concatenated [vec1|vec2]
#define TOPK   20
#define ALPHA  3.0f

union AFrag { v16h v; v8h h[2]; };

// ---------------------------------------------------------------------------
// Kernel 1: vec1/vec2 = tanh(3*(v@W.T + b)); pack f16 U=[vec1|vec2], Wm=[vec2|vec1]
// One block (128 threads) per row.
// ---------------------------------------------------------------------------
__global__ __launch_bounds__(128) void vec_kernel(
    const int* __restrict__ idx,
    const float* __restrict__ emb,
    const float* __restrict__ W1, const float* __restrict__ b1,
    const float* __restrict__ W2, const float* __restrict__ b2,
    float* __restrict__ vec1_out,          // f32, tail of d_out
    _Float16* __restrict__ U,              // [N, 256] f16
    _Float16* __restrict__ Wm)             // [N, 256] f16
{
    __shared__ float vrow[DIM];
    const int row = blockIdx.x;
    const int j   = threadIdx.x;

    const int src = idx[row];
    vrow[j] = emb[(size_t)src * DIM + j];
    __syncthreads();

    const float* w1r = W1 + (size_t)j * DIM;
    const float* w2r = W2 + (size_t)j * DIM;
    float acc1 = b1[j];
    float acc2 = b2[j];
#pragma unroll 8
    for (int k = 0; k < DIM; ++k) {
        float v = vrow[k];
        acc1 = fmaf(v, w1r[k], acc1);
        acc2 = fmaf(v, w2r[k], acc2);
    }
    float y1 = tanhf(ALPHA * acc1);
    float y2 = tanhf(ALPHA * acc2);

    vec1_out[(size_t)row * DIM + j] = y1;

    size_t ub = (size_t)row * KDIM;
    U[ub + j]        = (_Float16)y1;   // U  = [vec1 | vec2]
    U[ub + DIM + j]  = (_Float16)y2;
    Wm[ub + j]       = (_Float16)y2;   // Wm = [vec2 | vec1]
    Wm[ub + DIM + j] = (_Float16)y1;
}

// ---------------------------------------------------------------------------
// Kernel 2: adj = relu(tanh(3 * 0.5 * U @ Wm^T)), diagonal zeroed.
// Block tile 128x128, 8 waves, wave tile 32x64 (2x4 WMMA accumulators).
// K = 256 in 8 steps of 32 using v_wmma_f32_16x16x32_f16.
// ---------------------------------------------------------------------------
__global__ __launch_bounds__(256) void gemm_kernel(
    const _Float16* __restrict__ U,
    const _Float16* __restrict__ Wm,
    float* __restrict__ out)               // [N, N] adj
{
    const int lane  = threadIdx.x & 31;
    const int wid   = threadIdx.x >> 5;
    const int l15   = lane & 15;
    const int hsel  = lane >> 4;           // 0 for lanes 0-15, 1 for lanes 16-31

    const int bm = blockIdx.y * 128;
    const int bn = blockIdx.x * 128;
    const int wm = bm + (wid & 3) * 32;    // wave M base (2 sub-tiles of 16)
    const int wn = bn + (wid >> 2) * 64;   // wave N base (4 sub-tiles of 16)

    v8f acc[2][4];
#pragma unroll
    for (int r = 0; r < 2; ++r)
#pragma unroll
        for (int c = 0; c < 4; ++c)
            acc[r][c] = (v8f){};

#pragma unroll
    for (int kb = 0; kb < KDIM; kb += 32) {
        // A fragments: 16-bit A 16x32 layout.
        // lanes 0-15 : row M=l15, K = kb+{0..7} (VGPR0-3), kb+{16..23} (VGPR4-7)
        // lanes 16-31: row M=l15, K = kb+{8..15},           kb+{24..31}
        AFrag a[2];
#pragma unroll
        for (int r = 0; r < 2; ++r) {
            const _Float16* ap = U + (size_t)(wm + 16 * r + l15) * KDIM + kb + hsel * 8;
            a[r].h[0] = *(const v8h*)(ap);        // 16B aligned
            a[r].h[1] = *(const v8h*)(ap + 16);   // +32B
        }
        // B fragments: 16-bit B 32x16 layout.
        // lanes 0-15 : col N=l15, K = kb+{0..7},{16..23} ; lanes 16-31: col N=l15, K = kb+{8..15},{24..31}
        v16h b[4];
#pragma unroll
        for (int c = 0; c < 4; ++c) {
            const _Float16* bp = Wm + (size_t)(wn + 16 * c + l15) * KDIM + kb + hsel * 8;
            AFrag bf;
            bf.h[0] = *(const v8h*)(bp);
            bf.h[1] = *(const v8h*)(bp + 16);
            b[c] = bf.v;
        }
#pragma unroll
        for (int r = 0; r < 2; ++r)
#pragma unroll
            for (int c = 0; c < 4; ++c)
                acc[r][c] = __builtin_amdgcn_wmma_f32_16x16x32_f16(
                    false, a[r].v, false, b[c], (short)0, acc[r][c], false, false);
    }

    // Epilogue: C/D 16x16 layout -> element (vgpr p, lane): M = p + 8*hsel, N = l15
#pragma unroll
    for (int r = 0; r < 2; ++r) {
#pragma unroll
        for (int c = 0; c < 4; ++c) {
            int colg = wn + 16 * c + l15;
#pragma unroll
            for (int p = 0; p < 8; ++p) {
                int rowg = wm + 16 * r + p + 8 * hsel;
                float v = 0.5f * acc[r][c][p];
                if (rowg == colg) v = 0.0f;        // remove diagonal
                float t = tanhf(ALPHA * v);
                out[(size_t)rowg * N_ROWS + colg] = t > 0.0f ? t : 0.0f;  // relu
            }
        }
    }
}

// ---------------------------------------------------------------------------
// Kernel 3: exact top-K per row via radix select on float bits (all vals >= 0),
// then rewrite the row keeping exactly K entries.
// One block (256 threads) per row; row staged in LDS (32KB).
// ---------------------------------------------------------------------------
__global__ __launch_bounds__(256) void topk_kernel(float* __restrict__ out)
{
    __shared__ float row[N_ROWS];
    __shared__ int cnt;
    __shared__ int claimed;

    const int tid = threadIdx.x;
    float* rp = out + (size_t)blockIdx.x * N_ROWS;

    for (int i = tid; i < N_ROWS; i += 256) row[i] = rp[i];
    __syncthreads();

    // Radix select: largest t with count(bits >= t) >= K  ==> t = bits of K-th largest.
    unsigned t = 0u;
    for (int b = 31; b >= 0; --b) {
        unsigned cand = t | (1u << b);
        if (tid == 0) cnt = 0;
        __syncthreads();
        int local = 0;
        for (int i = tid; i < N_ROWS; i += 256)
            local += (__float_as_uint(row[i]) >= cand) ? 1 : 0;
        atomicAdd(&cnt, local);
        __syncthreads();
        if (cnt >= TOPK) t = cand;
        __syncthreads();
    }

    // Count strictly greater than t (always < K); boundary elems fill the rest.
    if (tid == 0) { cnt = 0; claimed = 0; }
    __syncthreads();
    int local = 0;
    for (int i = tid; i < N_ROWS; i += 256)
        local += (__float_as_uint(row[i]) > t) ? 1 : 0;
    atomicAdd(&cnt, local);
    __syncthreads();
    const int need = TOPK - cnt;

    for (int i = tid; i < N_ROWS; i += 256) {
        float v = row[i];
        unsigned bv = __float_as_uint(v);
        float o = 0.0f;
        if (bv > t) {
            o = v;
        } else if (bv == t) {
            int slot = atomicAdd(&claimed, 1);
            if (slot < need) o = v;
        }
        rp[i] = o;
    }
}

// ---------------------------------------------------------------------------
extern "C" void kernel_launch(void* const* d_in, const int* in_sizes, int n_in,
                              void* d_out, int out_size, void* d_ws, size_t ws_size,
                              hipStream_t stream) {
    const int*   idx = (const int*)d_in[0];
    const float* emb = (const float*)d_in[1];
    const float* W1  = (const float*)d_in[2];
    const float* b1  = (const float*)d_in[3];
    const float* W2  = (const float*)d_in[4];
    const float* b2  = (const float*)d_in[5];

    float* adj_out  = (float*)d_out;                                   // [N,N]
    float* vec1_out = (float*)d_out + (size_t)N_ROWS * N_ROWS;         // [N,D]

    _Float16* U  = (_Float16*)d_ws;                                    // [N,256]
    _Float16* Wm = U + (size_t)N_ROWS * KDIM;                          // [N,256]

    vec_kernel<<<N_ROWS, 128, 0, stream>>>(idx, emb, W1, b1, W2, b2,
                                           vec1_out, U, Wm);

    dim3 ggrid(N_ROWS / 128, N_ROWS / 128);   // 64 x 64
    gemm_kernel<<<ggrid, 256, 0, stream>>>(U, Wm, adj_out);

    topk_kernel<<<N_ROWS, 256, 0, stream>>>(adj_out);
}
